// Self_Attention_25383256719680
// MI455X (gfx1250) — matrix operand order repaired and checked by reading, not compiled, not run on hardware
//
#include <hip/hip_runtime.h>
#include <hip/hip_bf16.h>

#define BB  2
#define LL  2048
#define DD  1024
#define HH  16
#define HDD 64

typedef __attribute__((ext_vector_type(16))) __bf16 v16bf;
typedef __attribute__((ext_vector_type(8)))  float  v8f;

union BF16Frag { v16bf v; unsigned short u[16]; };
union F32x8   { v8f v; float f[8]; };

#define LOG2E 1.4426950408889634f

// Split f32 into hi+lo bf16 (truncation; lo captures residual -> ~16 mantissa bits)
__device__ inline void splitf(float x, unsigned short &hi, unsigned short &lo) {
  unsigned int uu = __float_as_uint(x) & 0xffff0000u;
  hi = (unsigned short)(uu >> 16);
  float r = x - __uint_as_float(uu);
  lo = (unsigned short)(__float_as_uint(r) >> 16);
}

__device__ inline v8f wmma_bf16(v16bf a, v16bf b, v8f c) {
  // (neg_a, A, neg_b, B, c_mod, C, reuse_a, reuse_b)
  return __builtin_amdgcn_wmma_f32_16x16x32_bf16(false, a, false, b, (short)0, c,
                                                 false, false);
}

// 3-product bf16x2 emulation: Ah*Bh + Ah*Bl + Al*Bh
__device__ inline v8f wmma3(v16bf ah, v16bf al, v16bf bh, v16bf bl, v8f c) {
  c = wmma_bf16(ah, bh, c);
  c = wmma_bf16(ah, bl, c);
  c = wmma_bf16(al, bh, c);
  return c;
}

// A-matrix (16x32) fragment from pre-split bf16, row-major [M][K]. lane&15 = m.
// Low-half lanes: K = k0+[0..7], k0+[16..23]; high half +8 (ISA 7.12.2). Two b128 loads.
__device__ inline v16bf ldA(const unsigned short* __restrict__ src, int ld, int lane, int k0) {
  const unsigned short* p = src + (long)(lane & 15) * ld + k0 + ((lane >> 4) << 3);
  BF16Frag f;
  *(uint4*)&f.u[0] = *(const uint4*)p;
  *(uint4*)&f.u[8] = *(const uint4*)(p + 16);
  return f.v;
}

// B-matrix (32x16) fragment where B[k][n] = src[n][k], src row-major bf16 [N][K].
// lane&15 = n; low half K = k0+[0..7], k0+[16..23]; high half +8. Two b128 loads.
__device__ inline v16bf ldB(const unsigned short* __restrict__ src, int ld, int lane, int k0) {
  const unsigned short* p = src + (long)(lane & 15) * ld + k0 + ((lane >> 4) << 3);
  BF16Frag f;
  *(uint4*)&f.u[0] = *(const uint4*)p;
  *(uint4*)&f.u[8] = *(const uint4*)(p + 16);
  return f.v;
}

// A-matrix fragment from LDS bf16, row-major [16][32]
__device__ inline v16bf ldA_lds32(const unsigned short* src, int lane) {
  const unsigned short* p = src + ((lane & 15) << 5) + ((lane >> 4) << 3);
  BF16Frag f;
  *(uint4*)&f.u[0] = *(const uint4*)p;
  *(uint4*)&f.u[8] = *(const uint4*)(p + 16);
  return f.v;
}

// ---------------- Kernel 0: elementwise f32 -> bf16 hi/lo pre-split ----------------
__global__ __launch_bounds__(256) void k_split(const float* __restrict__ src,
    unsigned short* __restrict__ dh, unsigned short* __restrict__ dl, int n4)
{
  const int i = blockIdx.x * 256 + threadIdx.x;
  if (i >= n4) return;
  const float4 f = ((const float4*)src)[i];
  ushort4 h, l;
  splitf(f.x, h.x, l.x); splitf(f.y, h.y, l.y);
  splitf(f.z, h.z, l.z); splitf(f.w, h.w, l.w);
  ((ushort4*)dh)[i] = h;
  ((ushort4*)dl)[i] = l;
}

// ---------------- Kernel 1: fused QKV projection ----------------
// 256 blocks; x row-tile staged in LDS once; each wave produces q,k,v for one n0 tile.
__global__ __launch_bounds__(256) void k_qkv(
    const unsigned short* __restrict__ xh, const unsigned short* __restrict__ xl,
    const unsigned short* __restrict__ Wah, const unsigned short* __restrict__ Wal, // [q|k|v|o]
    float* __restrict__ kout, float* __restrict__ vout,
    unsigned short* __restrict__ qsh, unsigned short* __restrict__ qsl,
    unsigned short* __restrict__ ksh, unsigned short* __restrict__ ksl,
    unsigned short* __restrict__ vth, unsigned short* __restrict__ vtl)
{
  __shared__ __align__(16) unsigned short Axh[16 * DD];   // 32 KB
  __shared__ __align__(16) unsigned short Axl[16 * DD];   // 32 KB
  const int tid = threadIdx.x, lane = tid & 31, wave = tid >> 5;
  const long r0 = (long)blockIdx.x * 16;
  const int b = (int)(r0 / LL), l0 = (int)(r0 % LL);

  {
    const uint4* sh4 = (const uint4*)(xh + r0 * DD);
    const uint4* sl4 = (const uint4*)(xl + r0 * DD);
    uint4* dh4 = (uint4*)Axh;
    uint4* dl4 = (uint4*)Axl;
    for (int i = tid; i < 16 * DD / 8; i += 256) { dh4[i] = sh4[i]; dl4[i] = sl4[i]; }
  }
  __syncthreads();

  const unsigned short* wqh = Wah;
  const unsigned short* wql = Wal;
  const unsigned short* wkh = Wah + (long)DD * DD;
  const unsigned short* wkl = Wal + (long)DD * DD;
  const unsigned short* wvh = Wah + (long)2 * DD * DD;
  const unsigned short* wvl = Wal + (long)2 * DD * DD;

  for (int t = wave; t < 64; t += 8) {
    asm volatile("" ::: "memory");      // keep fragment loads inside the loop (no LICM spill)
    const int n0 = t << 4;
    const long woff = (long)n0 * DD;
    F32x8 cq, ck, cv;
#pragma unroll
    for (int i = 0; i < 8; ++i) { cq.f[i] = 0.f; ck.f[i] = 0.f; cv.f[i] = 0.f; }

    for (int k0 = 0; k0 < DD; k0 += 32) {
      const v16bf ah = ldA(Axh, DD, lane, k0);
      const v16bf al = ldA(Axl, DD, lane, k0);
      cq.v = wmma3(ah, al, ldB(wqh + woff, DD, lane, k0), ldB(wql + woff, DD, lane, k0), cq.v);
      ck.v = wmma3(ah, al, ldB(wkh + woff, DD, lane, k0), ldB(wkl + woff, DD, lane, k0), ck.v);
      cv.v = wmma3(ah, al, ldB(wvh + woff, DD, lane, k0), ldB(wvl + woff, DD, lane, k0), cv.v);
    }

    const int n = lane & 15, mb = (lane >> 4) << 3;
    const int j = n0 + n;
    const int h = j >> 6, hd = j & 63;
#pragma unroll
    for (int v = 0; v < 8; ++v) {
      const int m = mb + v;
      const long idx = (((long)b * HH + h) * LL + (l0 + m)) * HDD + hd;
      const long ti  = (((long)b * HH + h) * HDD + hd) * LL + (l0 + m);
      unsigned short hb, lb;
      splitf(cq.f[v], hb, lb); qsh[idx] = hb; qsl[idx] = lb;
      splitf(ck.f[v], hb, lb); kout[idx] = ck.f[v]; ksh[idx] = hb; ksl[idx] = lb;
      splitf(cv.f[v], hb, lb); vout[idx] = cv.f[v]; vth[ti] = hb; vtl[ti] = lb;
    }
  }
}

// ---------------- Kernel 2: causal flash attention ----------------
// One wave per (b,h,16-query tile); 32 keys/iter; online softmax in exp2 domain.
__global__ __launch_bounds__(256) void k_attn(
    const unsigned short* __restrict__ qsh, const unsigned short* __restrict__ qsl,
    const unsigned short* __restrict__ ksh, const unsigned short* __restrict__ ksl,
    const unsigned short* __restrict__ vth, const unsigned short* __restrict__ vtl,
    const float* __restrict__ pe,
    unsigned short* __restrict__ osh, unsigned short* __restrict__ osl)
{
  __shared__ __align__(16) unsigned short Ph[8][512];
  __shared__ __align__(16) unsigned short Pl[8][512];
  const int tid = threadIdx.x, lane = tid & 31, wave = tid >> 5;
  const int unit = blockIdx.x * 8 + wave;          // 0..4095
  const int qt = unit & 127;
  const int h  = (unit >> 7) & 15;
  const int b  = unit >> 11;
  const int qi0 = qt << 4;

  const long bh = (long)b * HH + h;
  const unsigned short* qbh = qsh + (bh * LL + qi0) * HDD;
  const unsigned short* qbl = qsl + (bh * LL + qi0) * HDD;
  const unsigned short* kbh = ksh + bh * LL * HDD;
  const unsigned short* kbl = ksl + bh * LL * HDD;
  const unsigned short* vbh = vth + bh * HDD * LL;
  const unsigned short* vbl = vtl + bh * HDD * LL;
  const float* pebase = pe + ((long)h * LL + qi0) * LL;

  const v16bf qh0 = ldA(qbh, HDD, lane, 0),  ql0 = ldA(qbl, HDD, lane, 0);
  const v16bf qh1 = ldA(qbh, HDD, lane, 32), ql1 = ldA(qbl, HDD, lane, 32);

  F32x8 o[4];
#pragma unroll
  for (int t = 0; t < 4; ++t)
#pragma unroll
    for (int i = 0; i < 8; ++i) o[t].f[i] = 0.f;
  float mrun[8], srun[8];
#pragma unroll
  for (int i = 0; i < 8; ++i) { mrun[i] = -1e30f; srun[i] = 0.f; }

  const int n = lane & 15, mb = (lane >> 4) << 3;
  const float escale = 0.125f * LOG2E;    // (1/sqrt(64)) * log2(e)
  unsigned short* php = Ph[wave];
  unsigned short* plp = Pl[wave];

  const int nkb = (qi0 >> 5) + 1;         // causal: key blocks up to the diagonal
  for (int kb = 0; kb < nkb; ++kb) {
    const int key0 = kb << 5;
    F32x8 s0, s1;
#pragma unroll
    for (int i = 0; i < 8; ++i) { s0.f[i] = 0.f; s1.f[i] = 0.f; }

    s0.v = wmma3(qh0, ql0, ldB(kbh + (long)key0 * HDD, HDD, lane, 0),
                           ldB(kbl + (long)key0 * HDD, HDD, lane, 0),  s0.v);
    s0.v = wmma3(qh1, ql1, ldB(kbh + (long)key0 * HDD, HDD, lane, 32),
                           ldB(kbl + (long)key0 * HDD, HDD, lane, 32), s0.v);
    s1.v = wmma3(qh0, ql0, ldB(kbh + (long)(key0 + 16) * HDD, HDD, lane, 0),
                           ldB(kbl + (long)(key0 + 16) * HDD, HDD, lane, 0),  s1.v);
    s1.v = wmma3(qh1, ql1, ldB(kbh + (long)(key0 + 16) * HDD, HDD, lane, 32),
                           ldB(kbl + (long)(key0 + 16) * HDD, HDD, lane, 32), s1.v);

#pragma unroll
    for (int v = 0; v < 8; ++v) {
      const int m = mb + v;
      const int qrow = qi0 + m;
      // exp2-domain scores: e = (s*scale + pe) * log2(e)
      float e0 = fmaf(s0.f[v], escale, pebase[(long)m * LL + key0 + n] * LOG2E);
      float e1 = fmaf(s1.f[v], escale, pebase[(long)m * LL + key0 + 16 + n] * LOG2E);
      if (key0 + n > qrow)      e0 = -1e9f;   // analytic causal mask
      if (key0 + 16 + n > qrow) e1 = -1e9f;
      float mx = fmaxf(e0, e1);
#pragma unroll
      for (int off = 8; off >= 1; off >>= 1)
        mx = fmaxf(mx, __shfl_xor(mx, off, 32));   // stays within 16-lane half
      const float nm    = fmaxf(mrun[v], mx);
      const float alpha = exp2f(mrun[v] - nm);
      const float p0 = exp2f(e0 - nm);
      const float p1 = exp2f(e1 - nm);
      float rs = p0 + p1;
#pragma unroll
      for (int off = 8; off >= 1; off >>= 1)
        rs += __shfl_xor(rs, off, 32);
      srun[v] = srun[v] * alpha + rs;
      mrun[v] = nm;
#pragma unroll
      for (int t = 0; t < 4; ++t) o[t].f[v] *= alpha;
      unsigned short hb, lb;
      splitf(p0, hb, lb); php[m * 32 + n]      = hb; plp[m * 32 + n]      = lb;
      splitf(p1, hb, lb); php[m * 32 + 16 + n] = hb; plp[m * 32 + 16 + n] = lb;
    }

    const v16bf pah = ldA_lds32(php, lane);  // same-wave DS ops are in-order (ISA 7.3)
    const v16bf pal = ldA_lds32(plp, lane);
#pragma unroll
    for (int t = 0; t < 4; ++t) {
      o[t].v = wmma3(pah, pal,
                     ldB(vbh + (long)(t * 16) * LL, LL, lane, key0),
                     ldB(vbl + (long)(t * 16) * LL, LL, lane, key0), o[t].v);
    }
  }

#pragma unroll
  for (int v = 0; v < 8; ++v) {
    const int m = mb + v;
    const float inv = 1.f / srun[v];
#pragma unroll
    for (int t = 0; t < 4; ++t) {
      const long oi = ((long)b * LL + qi0 + m) * DD + h * HDD + t * 16 + n;
      unsigned short hb, lb;
      splitf(o[t].f[v] * inv, hb, lb);
      osh[oi] = hb; osl[oi] = lb;
    }
  }
}

// ---------------- Kernel 3: output projection ----------------
__global__ __launch_bounds__(256) void k_oproj(
    const unsigned short* __restrict__ osh, const unsigned short* __restrict__ osl,
    const unsigned short* __restrict__ Wah, const unsigned short* __restrict__ Wal,
    float* __restrict__ out)
{
  __shared__ __align__(16) unsigned short Aoh[16 * DD];   // 32 KB
  __shared__ __align__(16) unsigned short Aol[16 * DD];   // 32 KB
  const int tid = threadIdx.x, lane = tid & 31, wave = tid >> 5;
  const long r0 = (long)blockIdx.x * 16;

  {
    const uint4* sh4 = (const uint4*)(osh + r0 * DD);
    const uint4* sl4 = (const uint4*)(osl + r0 * DD);
    uint4* dh4 = (uint4*)Aoh;
    uint4* dl4 = (uint4*)Aol;
    for (int i = tid; i < 16 * DD / 8; i += 256) { dh4[i] = sh4[i]; dl4[i] = sl4[i]; }
  }
  __syncthreads();

  const unsigned short* woh = Wah + (long)3 * DD * DD;   // Wo slot
  const unsigned short* wol = Wal + (long)3 * DD * DD;

  for (int t = wave; t < 64; t += 8) {
    asm volatile("" ::: "memory");      // keep fragment loads inside the loop
    const int n0 = t << 4;
    F32x8 c;
#pragma unroll
    for (int i = 0; i < 8; ++i) c.f[i] = 0.f;
    for (int k0 = 0; k0 < DD; k0 += 32) {
      c.v = wmma3(ldA(Aoh, DD, lane, k0), ldA(Aol, DD, lane, k0),
                  ldB(woh + (long)n0 * DD, DD, lane, k0),
                  ldB(wol + (long)n0 * DD, DD, lane, k0), c.v);
    }
    const int n = lane & 15, mb = (lane >> 4) << 3;
#pragma unroll
    for (int v = 0; v < 8; ++v)
      out[(r0 + mb + v) * DD + n0 + n] = c.f[v];
  }
}

extern "C" void kernel_launch(void* const* d_in, const int* in_sizes, int n_in,
                              void* d_out, int out_size, void* d_ws, size_t ws_size,
                              hipStream_t stream)
{
  (void)in_sizes; (void)n_in; (void)out_size; (void)ws_size;
  const float* x  = (const float*)d_in[0];
  const float* pe = (const float*)d_in[1];
  /* d_in[2] = mask: unused, causality computed analytically */
  const float* Wq = (const float*)d_in[3];
  const float* Wk = (const float*)d_in[4];
  const float* Wv = (const float*)d_in[5];
  const float* Wo = (const float*)d_in[6];

  float* proj = (float*)d_out;                       // [B,L,D]
  float* kout = proj + (long)BB * LL * DD;           // [B,H,L,HD]
  float* vout = kout + (long)BB * HH * LL * HDD;     // [B,H,L,HD]

  const long NX = (long)BB * LL * DD;                // 4M elements
  const long NW = (long)DD * DD;                     // 1M elements
  unsigned short* w = (unsigned short*)d_ws;
  unsigned short* xh  = w; w += NX;
  unsigned short* xl  = w; w += NX;
  unsigned short* Wah = w; w += 4 * NW;              // [Wq|Wk|Wv|Wo] hi
  unsigned short* Wal = w; w += 4 * NW;              // [Wq|Wk|Wv|Wo] lo
  unsigned short* qsh = w; w += NX;                  // [B,H,L,HD]
  unsigned short* qsl = w; w += NX;
  unsigned short* ksh = w; w += NX;                  // [B,H,L,HD]
  unsigned short* ksl = w; w += NX;
  unsigned short* vth = w; w += NX;                  // [B,H,HD,L]
  unsigned short* vtl = w; w += NX;
  unsigned short* osh = w; w += NX;                  // [B,L,D]
  unsigned short* osl = w; w += NX;                  // total 96 MB

  // Pre-split inputs once (elementwise, memory-bound, trivial cost)
  k_split<<<(int)(NX / 4 / 256), 256, 0, stream>>>(x, xh, xl, (int)(NX / 4));
  const float* Ws[4] = { Wq, Wk, Wv, Wo };
  for (int i = 0; i < 4; ++i)
    k_split<<<(int)(NW / 4 / 256), 256, 0, stream>>>(Ws[i], Wah + i * NW, Wal + i * NW,
                                                     (int)(NW / 4));

  k_qkv  <<<256, 256, 0, stream>>>(xh, xl, Wah, Wal, kout, vout,
                                   qsh, qsl, ksh, ksl, vth, vtl);
  k_attn <<<512, 256, 0, stream>>>(qsh, qsl, ksh, ksl, vth, vtl, pe, osh, osl);
  k_oproj<<<256, 256, 0, stream>>>(osh, osl, Wah, Wal, proj);
}
